// RNN_48558900248904
// MI455X (gfx1250) — matrix operand order repaired and checked by reading, not compiled, not run on hardware
//
#include <hip/hip_runtime.h>
#include <hip/hip_bf16.h>
#include <math.h>

typedef __attribute__((ext_vector_type(16))) _Float16 v16h;
typedef __attribute__((ext_vector_type(8)))  _Float16 v8h;
typedef __attribute__((ext_vector_type(8)))  float    v8f;

#define B_    8192
#define T_    200
#define DD_   8
#define NS_   4
#define V_    1000
#define E_    32
#define H_    256
#define KIN_  136   // DD + NS*E
#define KP_   160   // KIN padded to multiple of 32
#define BT_   32    // batch rows per workgroup
#define LDX_  168   // xbuf LDS row stride (halves), conflict-free
#define LDH_  264   // h/x LDS row stride (halves), conflict-free

#define WL_OFF  0
#define WIH_OFF (256 * KP_)            // 40960 halves
#define WHH_OFF (WIH_OFF + 4*H_*H_)    // + 262144
#define WTOT_   (WHH_OFF + 4*H_*H_)    // 565248 halves total

// ---------------- weight prep: fp32 -> f16, Wl zero-padded to K=160 ----------
__global__ void prep_weights(const float* __restrict__ Wl,
                             const float* __restrict__ Wih,
                             const float* __restrict__ Whh,
                             _Float16* __restrict__ wb) {
  int i = blockIdx.x * blockDim.x + threadIdx.x;
  if (i < 256 * KP_) {
    int n = i / KP_, k = i % KP_;
    wb[WL_OFF + i] = (_Float16)(k < KIN_ ? Wl[n * KIN_ + k] : 0.0f);
    return;
  }
  i -= 256 * KP_;
  if (i < 4 * H_ * H_) { wb[WIH_OFF + i] = (_Float16)Wih[i]; return; }
  i -= 4 * H_ * H_;
  if (i < 4 * H_ * H_) { wb[WHH_OFF + i] = (_Float16)Whh[i]; return; }
}

// A fragment (16x32 f16) from LDS, contiguous k order (matches B):
// lane L (0-15): M=L, k = kb+{0..15};
// lane L+16:     M=L, k = kb+16+{0..15}.
static __device__ __forceinline__ v16h load_a_lds(const _Float16* src, int ld,
                                                  int m0, int kb, int lane) {
  int m = m0 + (lane & 15);
  const _Float16* p = src + m * ld + kb + ((lane >> 4) << 4);
  union { v16h v; v8h h[2]; } u;
  u.h[0] = *(const v8h*)(p);
  u.h[1] = *(const v8h*)(p + 8);
  return u.v;
}

// acc[mt][j] += A(m-tile mt) x W^T(n-tile j) over nk k-chunks of 32.
// Each lane's B fragment is one contiguous, 32B-aligned run of a weight row;
// every B fragment feeds 2 WMMAs (both M tiles) -> half the weight traffic.
static __device__ __forceinline__ void gemm_accum(v8f acc[2][2], const _Float16* a_lds,
                                                  int ld,
                                                  const _Float16* __restrict__ wmat,
                                                  const int* rowoff, int lane, int nk) {
#pragma unroll
  for (int kk = 0; kk < 8; ++kk) {
    if (kk >= nk) break;
    v16h a0 = load_a_lds(a_lds, ld, 0,  kk * 32, lane);
    v16h a1 = load_a_lds(a_lds, ld, 16, kk * 32, lane);
#pragma unroll
    for (int j = 0; j < 2; ++j) {
      v16h b = *(const v16h*)(wmat + rowoff[j] + kk * 32);
      acc[0][j] = __builtin_amdgcn_wmma_f32_16x16x32_f16(
          false, a0, false, b, (short)0, acc[0][j], false, false);
      acc[1][j] = __builtin_amdgcn_wmma_f32_16x16x32_f16(
          false, a1, false, b, (short)0, acc[1][j], false, false);
    }
  }
}

__global__ __launch_bounds__(256)
void rnn_fused(const float* __restrict__ dense, const int* __restrict__ sparse,
               const int* __restrict__ lengths, const float* __restrict__ emb,
               const float* __restrict__ bl, const float* __restrict__ bih,
               const float* __restrict__ bhh, const float* __restrict__ Wout,
               const float* __restrict__ bout, const _Float16* __restrict__ wb,
               float* __restrict__ out) {
  __shared__ __attribute__((aligned(32))) _Float16 s_xb[BT_ * LDX_]; // input features
  __shared__ __attribute__((aligned(32))) _Float16 s_xh[BT_ * LDH_]; // input-layer act
  __shared__ __attribute__((aligned(32))) _Float16 s_h0[BT_ * LDH_]; // layer0 hidden
  __shared__ __attribute__((aligned(32))) _Float16 s_h1[BT_ * LDH_]; // layer1 hidden
  __shared__ int s_idx[BT_ * NS_];
  __shared__ int s_len[BT_];

  const int tid   = threadIdx.x;
  const int lane  = tid & 31;
  const int wid   = tid >> 5;           // 0..7 -> owns n-tiles {2*wid, 2*wid+1}
  const int nl    = lane & 15;          // B-frag column within tile / C col
  const int khalf = lane >> 4;          // 0/1
  const int brow0 = blockIdx.x * BT_;

  // fill-phase coordinates: 256 threads = 32 rows x 8 slices
  const int fr = tid >> 3;              // 0..31 batch row within tile
  const int fs = tid & 7;               // 0..7 slice

  const _Float16* wlT = wb + WL_OFF;
  const _Float16* wih = wb + WIH_OFF;   // [L][2][H][H]
  const _Float16* whh = wb + WHH_OFF;

  // per-lane weight-row offsets + hoisted biases (2 n-tiles per wave)
  int rowoffW[2], rowoffL[2], col[2];
  float blv[2], bsum0[2][2], bsum1[2][2];
#pragma unroll
  for (int j = 0; j < 2; ++j) {
    int nt = wid * 2 + j;
    col[j]     = nt * 16 + nl;
    rowoffW[j] = col[j] * H_ + khalf * 16;
    rowoffL[j] = col[j] * KP_ + khalf * 16;
    blv[j]     = bl[col[j]];
#pragma unroll
    for (int p = 0; p < 2; ++p) {
      bsum0[p][j] = bih[(0 * 2 + p) * H_ + col[j]] + bhh[(0 * 2 + p) * H_ + col[j]];
      bsum1[p][j] = bih[(1 * 2 + p) * H_ + col[j]] + bhh[(1 * 2 + p) * H_ + col[j]];
    }
  }

  // init hidden state, lengths, and the K-pad region of s_xb (written once;
  // the per-step fill only touches columns < KIN_)
  for (int i = tid; i < BT_ * LDH_; i += 256) { s_h0[i] = (_Float16)0.f; s_h1[i] = (_Float16)0.f; }
  for (int i = tid; i < BT_ * (KP_ - KIN_); i += 256) {
    int r = i / (KP_ - KIN_), c = KIN_ + i % (KP_ - KIN_);
    s_xb[r * LDX_ + c] = (_Float16)0.f;
  }
  if (tid < BT_) s_len[tid] = lengths[brow0 + tid];
  __syncthreads();

  for (int t = 0; t < T_; ++t) {
    const int p = t & 1;
    const _Float16* WihP = wih + p * (H_ * H_);
    const _Float16* WhhP = whh + p * (H_ * H_);

    // ---- gather sparse ids for this step
    if (tid < BT_ * NS_) {
      int r = tid >> 2, f = tid & 3;
      s_idx[tid] = sparse[(brow0 + r) * (T_ * NS_) + t * NS_ + f];
    }
    __syncthreads();

    // ---- build x_in = [dense | emb0..emb3] as f16 in LDS (vectorized)
    // dense: thread (fr, fs) covers exactly one of 32x8 elements
    s_xb[fr * LDX_ + fs] =
        (_Float16)dense[(brow0 + fr) * (T_ * DD_) + t * DD_ + fs];
    // embeddings: thread (fr, fs) covers 16 consecutive f16 of row fr
    {
      const int c0 = fs * 16;                 // 0..112, within one 32-wide feature
      const int f  = c0 >> 5;
      const int e0 = c0 & 31;                 // 0 or 16
      const float* erow = emb + (f * V_ + s_idx[fr * NS_ + f]) * E_ + e0;
      union { v8h v; _Float16 h[8]; } o0, o1;
#pragma unroll
      for (int q = 0; q < 8; ++q) {
        o0.h[q] = (_Float16)erow[q];
        o1.h[q] = (_Float16)erow[8 + q];
      }
      *(v8h*)&s_xb[fr * LDX_ + DD_ + c0]     = o0.v;
      *(v8h*)&s_xb[fr * LDX_ + DD_ + c0 + 8] = o1.v;
    }
    __syncthreads();

    // ---- input layer: x = relu(x_in @ Wl^T + bl)
    v8f acc[2][2];
#pragma unroll
    for (int mt = 0; mt < 2; ++mt)
#pragma unroll
      for (int j = 0; j < 2; ++j)
#pragma unroll
        for (int e = 0; e < 8; ++e) acc[mt][j][e] = blv[j];
    gemm_accum(acc, s_xb, LDX_, wlT, rowoffL, lane, 5);
#pragma unroll
    for (int mt = 0; mt < 2; ++mt)
#pragma unroll
      for (int j = 0; j < 2; ++j)
#pragma unroll
        for (int r = 0; r < 8; ++r) {
          int row = mt * 16 + r + khalf * 8;
          s_xh[row * LDH_ + col[j]] = (_Float16)fmaxf(acc[mt][j][r], 0.f);
        }
    __syncthreads();

    // ---- layer 0: h0 = mask ? tanh(x@Wih0^T + h0@Whh0^T + b) : h0
#pragma unroll
    for (int mt = 0; mt < 2; ++mt)
#pragma unroll
      for (int j = 0; j < 2; ++j)
#pragma unroll
        for (int e = 0; e < 8; ++e) acc[mt][j][e] = bsum0[p][j];
    gemm_accum(acc, s_xh, LDH_, WihP, rowoffW, lane, 8);
    gemm_accum(acc, s_h0, LDH_, WhhP, rowoffW, lane, 8);
    __syncthreads();  // all reads of s_h0 done before masked writes
#pragma unroll
    for (int mt = 0; mt < 2; ++mt)
#pragma unroll
      for (int j = 0; j < 2; ++j)
#pragma unroll
        for (int r = 0; r < 8; ++r) {
          int row = mt * 16 + r + khalf * 8;
          if (t < s_len[row]) s_h0[row * LDH_ + col[j]] = (_Float16)tanhf(acc[mt][j][r]);
        }
    __syncthreads();

    // ---- layer 1: h1 = mask ? tanh(h0@Wih1^T + h1@Whh1^T + b) : h1
#pragma unroll
    for (int mt = 0; mt < 2; ++mt)
#pragma unroll
      for (int j = 0; j < 2; ++j)
#pragma unroll
        for (int e = 0; e < 8; ++e) acc[mt][j][e] = bsum1[p][j];
    gemm_accum(acc, s_h0, LDH_, WihP + 2 * H_ * H_, rowoffW, lane, 8);
    gemm_accum(acc, s_h1, LDH_, WhhP + 2 * H_ * H_, rowoffW, lane, 8);
    __syncthreads();
#pragma unroll
    for (int mt = 0; mt < 2; ++mt)
#pragma unroll
      for (int j = 0; j < 2; ++j)
#pragma unroll
        for (int r = 0; r < 8; ++r) {
          int row = mt * 16 + r + khalf * 8;
          if (t < s_len[row]) s_h1[row * LDH_ + col[j]] = (_Float16)tanhf(acc[mt][j][r]);
        }
    __syncthreads();
  }

  // ---- head: sigmoid(h1 @ Wout^T + bout)
  if (tid < BT_) {
    float s = bout[0];
#pragma unroll 4
    for (int k = 0; k < H_; ++k) s += (float)s_h1[tid * LDH_ + k] * Wout[k];
    out[brow0 + tid] = 1.0f / (1.0f + expf(-s));
  }
}

extern "C" void kernel_launch(void* const* d_in, const int* in_sizes, int n_in,
                              void* d_out, int out_size, void* d_ws, size_t ws_size,
                              hipStream_t stream) {
  const float* dense   = (const float*)d_in[0];
  const int*   sparse  = (const int*)  d_in[1];
  const int*   lengths = (const int*)  d_in[2];
  const float* emb     = (const float*)d_in[3];
  const float* Wl      = (const float*)d_in[4];
  const float* bl      = (const float*)d_in[5];
  const float* Wih     = (const float*)d_in[6];
  const float* Whh     = (const float*)d_in[7];
  const float* bih     = (const float*)d_in[8];
  const float* bhh     = (const float*)d_in[9];
  const float* Wout    = (const float*)d_in[10];
  const float* bout    = (const float*)d_in[11];
  _Float16* wbuf = (_Float16*)d_ws;

  prep_weights<<<(WTOT_ + 255) / 256, 256, 0, stream>>>(Wl, Wih, Whh, wbuf);
  rnn_fused<<<B_ / BT_, 256, 0, stream>>>(dense, sparse, lengths, emb,
                                          bl, bih, bhh, Wout, bout, wbuf,
                                          (float*)d_out);
}
